// Baseline_GNN_25520695673361
// MI455X (gfx1250) — matrix operand order repaired and checked by reading, not compiled, not run on hardware
//
#include <hip/hip_runtime.h>
#include <hip/hip_bf16.h>

#define HD 64            // hidden dim
#define GD 64            // graphs
#define THREADS 256

typedef __attribute__((ext_vector_type(16))) _Float16 v16h;
typedef __attribute__((ext_vector_type(8)))  float    v8f;

// ---------------------------------------------------------------------------
// generic fill
// ---------------------------------------------------------------------------
__global__ void k_fill(float* __restrict__ p, float v, int n) {
    int i = blockIdx.x * blockDim.x + threadIdx.x;
    if (i < n) p[i] = v;
}

// ---------------------------------------------------------------------------
// degree: deg[dst] += 1 per edge (deg pre-filled with 1.0 for self loop)
// ---------------------------------------------------------------------------
__global__ void k_deg(const int* __restrict__ ei, float* __restrict__ deg, int nE) {
    int e = blockIdx.x * blockDim.x + threadIdx.x;
    if (e >= nE) return;
    atomicAdd(&deg[ei[nE + e]], 1.0f);
}

__global__ void k_rsqrt_inplace(float* __restrict__ p, int n) {
    int i = blockIdx.x * blockDim.x + threadIdx.x;
    if (i < n) p[i] = rsqrtf(p[i]);
}

// ---------------------------------------------------------------------------
// Y[N,64] = X[N,64] @ W[64,64]   (fp32 in/out, f16 WMMA accumulate-in-f32)
// One 16-row output tile per wave, 8 waves per block.
// B staged in LDS in WMMA B layout: N = lane&15, element j = K in same order as A.
// A per ISA 16-bit A layout: lanes 0-15 K{0..7,16..23}, lanes 16-31 K{8..15,24..31}.
// C per ISA: lanes 0-15 -> M=r, lanes 16-31 -> M=r+8, N = lane&15.
// ---------------------------------------------------------------------------
__global__ void k_gemm_h64_wmma(const float* __restrict__ X,
                                const float* __restrict__ W,
                                float* __restrict__ Y, int nrows) {
    __shared__ __align__(32) _Float16 Bp[2][4][32][16];  // [kb][nb][lane][j] = 8 KB
    int tid = threadIdx.x;

    // pack W (fp32 64x64) -> LDS f16 in WMMA-B tile layout
    {
        int kb   = tid >> 7;         // 0..1  (K block of 32)
        int nb   = (tid >> 5) & 3;   // 0..3  (N block of 16)
        int lane = tid & 31;         // N = lane&15, K half = lane>>4
        int k    = (kb << 5) + ((lane >> 4) << 3);
        const float* wcol = W + (nb << 4) + (lane & 15);
#pragma unroll
        for (int j = 0; j < 16; ++j)
            Bp[kb][nb][lane][j] = (_Float16)wcol[(k + (j & 7) + ((j >> 3) << 4)) * HD];
    }
    __syncthreads();

    int wave = (blockIdx.x << 3) + (tid >> 5);
    if (wave * 16 >= nrows) return;              // wave-uniform

    int lane    = tid & 31;
    int m       = lane & 15;
    int hi      = lane >> 4;
    int rowBase = wave << 4;
    const float* xrow = X + (size_t)(rowBase + m) * HD;

    v8f acc[4] = {};
#pragma unroll
    for (int kb = 0; kb < 2; ++kb) {
        v16h a;
        int k0 = (kb << 5) + (hi << 3);
#pragma unroll
        for (int j = 0; j < 8; ++j) a[j]     = (_Float16)xrow[k0 + j];
#pragma unroll
        for (int j = 0; j < 8; ++j) a[8 + j] = (_Float16)xrow[k0 + 16 + j];
#pragma unroll
        for (int nb = 0; nb < 4; ++nb) {
            v16h b = *(const v16h*)&Bp[kb][nb][lane][0];
            acc[nb] = __builtin_amdgcn_wmma_f32_16x16x32_f16(
                false, a, false, b, (short)0, acc[nb], false, false);
        }
    }
#pragma unroll
    for (int nb = 0; nb < 4; ++nb) {
        int n = (nb << 4) + m;
#pragma unroll
        for (int r = 0; r < 8; ++r) {
            int row = rowBase + (hi << 3) + r;
            Y[(size_t)row * HD + n] = acc[nb][r];
        }
    }
}

// ---------------------------------------------------------------------------
// BatchNorm stats: stats[0:64]=sum, stats[64:128]=sum of squares (atomics)
// ---------------------------------------------------------------------------
__global__ void k_bn_stats(const float* __restrict__ Yv, float* __restrict__ stats, int n) {
    __shared__ float sh[2][THREADS];
    int col = threadIdx.x & 63;
    int rg  = threadIdx.x >> 6;          // 0..3
    float s = 0.f, s2 = 0.f;
    for (int r = blockIdx.x * 4 + rg; r < n; r += gridDim.x * 4) {
        float v = Yv[(size_t)r * HD + col];
        s += v; s2 += v * v;
    }
    sh[0][threadIdx.x] = s;
    sh[1][threadIdx.x] = s2;
    __syncthreads();
    if (rg == 0) {
        s  = sh[0][col] + sh[0][64 + col] + sh[0][128 + col] + sh[0][192 + col];
        s2 = sh[1][col] + sh[1][64 + col] + sh[1][128 + col] + sh[1][192 + col];
        atomicAdd(&stats[col], s);
        atomicAdd(&stats[64 + col], s2);
    }
}

__global__ void k_bn_apply(float* __restrict__ Yv, const float* __restrict__ stats,
                           const float* __restrict__ g, const float* __restrict__ beta,
                           int n, int relu) {
    int idx = blockIdx.x * blockDim.x + threadIdx.x;
    if (idx >= n * HD) return;
    int col = idx & 63;
    float inv_n = 1.0f / (float)n;
    float mean  = stats[col] * inv_n;
    float var   = stats[64 + col] * inv_n - mean * mean;
    float rs    = rsqrtf(var + 1e-5f);
    float v = (Yv[idx] - mean) * rs * g[col] + beta[col];
    if (relu) v = fmaxf(v, 0.f);
    Yv[idx] = v;
}

// ---------------------------------------------------------------------------
// GCN edge scatter: agg[dst] += hw[src] * dinv[src]*dinv[dst]  (4 thr/edge)
// ---------------------------------------------------------------------------
__global__ void k_gcn_scatter(const float* __restrict__ hw, const int* __restrict__ ei,
                              const float* __restrict__ dinv, float* __restrict__ agg, int nE) {
    int t = blockIdx.x * blockDim.x + threadIdx.x;
    int e = t >> 2;
    if (e >= nE) return;
    int part = t & 3;
    int s = ei[e];
    int d = ei[nE + e];
    float w = dinv[s] * dinv[d];
    const float* sp = hw  + (size_t)s * HD + (part << 4);
    float*       dp = agg + (size_t)d * HD + (part << 4);
#pragma unroll
    for (int c = 0; c < 16; ++c) atomicAdd(&dp[c], sp[c] * w);
}

// out = agg + hw*dinv^2 + bias + hin ; optional relu
__global__ void k_gcn_finish(const float* __restrict__ hin, const float* __restrict__ hw,
                             const float* __restrict__ agg, const float* __restrict__ dinv,
                             const float* __restrict__ bias, float* __restrict__ hout,
                             int n, int relu) {
    int idx = blockIdx.x * blockDim.x + threadIdx.x;
    if (idx >= n * HD) return;
    int row = idx >> 6, col = idx & 63;
    float di = dinv[row];
    float v = agg[idx] + hw[idx] * di * di + bias[col] + hin[idx];
    if (relu) v = fmaxf(v, 0.f);
    hout[idx] = v;
}

// ---------------------------------------------------------------------------
// pooling
// ---------------------------------------------------------------------------
__global__ void k_pool_accum(const float* __restrict__ hv, const int* __restrict__ batch,
                             float* __restrict__ pooled, float* __restrict__ counts, int n) {
    int idx = blockIdx.x * blockDim.x + threadIdx.x;
    if (idx >= n * HD) return;
    int row = idx >> 6, col = idx & 63;
    int g = batch[row];
    atomicAdd(&pooled[g * HD + col], hv[idx]);
    if (col == 0) atomicAdd(&counts[g], 1.0f);
}

// ---------------------------------------------------------------------------
// final head: mean, fc-relu, fc-relu, fc -> out[64,2]   (single block)
// ---------------------------------------------------------------------------
__global__ void k_head(const float* __restrict__ pooled, const float* __restrict__ counts,
                       const float* __restrict__ W0, const float* __restrict__ b0,
                       const float* __restrict__ W1, const float* __restrict__ b1,
                       const float* __restrict__ W2, const float* __restrict__ b2,
                       float* __restrict__ out) {
    __shared__ float z0[GD * HD];
    __shared__ float z1[GD * HD];
    int tid = threadIdx.x;
    for (int i = tid; i < GD * HD; i += THREADS) {
        int r = i >> 6;
        z0[i] = pooled[i] / fmaxf(counts[r], 1.0f);
    }
    __syncthreads();
    for (int i = tid; i < GD * HD; i += THREADS) {
        int r = i >> 6, c = i & 63;
        float acc = b0[c];
        for (int k = 0; k < HD; ++k) acc += z0[r * HD + k] * W0[k * HD + c];
        z1[i] = fmaxf(acc, 0.f);
    }
    __syncthreads();
    for (int i = tid; i < GD * HD; i += THREADS) {
        int r = i >> 6, c = i & 63;
        float acc = b1[c];
        for (int k = 0; k < HD; ++k) acc += z1[r * HD + k] * W1[k * HD + c];
        z0[i] = fmaxf(acc, 0.f);
    }
    __syncthreads();
    for (int i = tid; i < GD * 2; i += THREADS) {
        int r = i >> 1, c = i & 1;
        float acc = b2[c];
        for (int k = 0; k < HD; ++k) acc += z0[r * HD + k] * W2[k * 2 + c];
        out[i] = acc;
    }
}

// ---------------------------------------------------------------------------
extern "C" void kernel_launch(void* const* d_in, const int* in_sizes, int n_in,
                              void* d_out, int out_size, void* d_ws, size_t ws_size,
                              hipStream_t stream) {
    const float* x        = (const float*)d_in[0];
    const int*   ei       = (const int*)  d_in[1];
    const int*   batch    = (const int*)  d_in[2];
    const float* pre_W    = (const float*)d_in[3];
    const float* pre_g    = (const float*)d_in[5];   // pre_b cancels in BN
    const float* pre_beta = (const float*)d_in[6];
    const float* conv_W   = (const float*)d_in[7];
    const float* conv_b   = (const float*)d_in[8];
    const float* post_W   = (const float*)d_in[9];
    const float* post_g   = (const float*)d_in[11];  // post_b cancels in BN
    const float* post_beta= (const float*)d_in[12];
    const float* fW0 = (const float*)d_in[13];
    const float* fb0 = (const float*)d_in[14];
    const float* fW1 = (const float*)d_in[15];
    const float* fb1 = (const float*)d_in[16];
    const float* fW2 = (const float*)d_in[17];
    const float* fb2 = (const float*)d_in[18];
    float* out = (float*)d_out;

    const int N_ = in_sizes[0] / HD;
    const int E_ = in_sizes[1] / 2;
    const size_t NH = (size_t)N_ * HD;

    float* ws     = (float*)d_ws;
    float* h0     = ws;
    float* h1     = ws + NH;
    float* hw     = ws + 2 * NH;
    float* agg    = ws + 3 * NH;
    float* dinv   = ws + 4 * NH;          // deg -> dinv in place
    float* stats  = dinv + N_;            // 128 floats
    float* pooled = stats + 128;          // 64*64
    float* counts = pooled + GD * HD;     // 64

    const int ew   = (int)NH;                       // elementwise total
    const int ewG  = (ew + THREADS - 1) / THREADS;
    const int tileG = ((N_ / 16) + 7) / 8;          // 8 waves per block

    // degree & dinv
    k_fill<<<(N_ + THREADS - 1) / THREADS, THREADS, 0, stream>>>(dinv, 1.0f, N_);
    k_deg<<<(E_ + THREADS - 1) / THREADS, THREADS, 0, stream>>>(ei, dinv, E_);
    k_rsqrt_inplace<<<(N_ + THREADS - 1) / THREADS, THREADS, 0, stream>>>(dinv, N_);

    const float* cur = x;
    float* A = h0;
    float* B = h1;

    // pre-processing: Linear -> BN (-> ReLU except last); Linear bias cancels in BN
    for (int i = 0; i < 3; ++i) {
        k_gemm_h64_wmma<<<tileG, THREADS, 0, stream>>>(cur, pre_W + i * HD * HD, A, N_);
        k_fill<<<1, 128, 0, stream>>>(stats, 0.f, 128);
        k_bn_stats<<<512, THREADS, 0, stream>>>(A, stats, N_);
        k_bn_apply<<<ewG, THREADS, 0, stream>>>(A, stats, pre_g + i * HD, pre_beta + i * HD,
                                                N_, i != 2);
        cur = A; float* t = A; A = B; B = t;
    }

    // GCN stack with residual
    for (int i = 0; i < 6; ++i) {
        k_gemm_h64_wmma<<<tileG, THREADS, 0, stream>>>(cur, conv_W + i * HD * HD, hw, N_);
        k_fill<<<ewG, THREADS, 0, stream>>>(agg, 0.f, ew);
        k_gcn_scatter<<<(4 * E_ + THREADS - 1) / THREADS, THREADS, 0, stream>>>(hw, ei, dinv, agg, E_);
        k_gcn_finish<<<ewG, THREADS, 0, stream>>>(cur, hw, agg, dinv, conv_b + i * HD, A,
                                                  N_, i != 5);
        cur = A; float* t = A; A = B; B = t;
    }

    // post-processing
    for (int i = 0; i < 3; ++i) {
        k_gemm_h64_wmma<<<tileG, THREADS, 0, stream>>>(cur, post_W + i * HD * HD, A, N_);
        k_fill<<<1, 128, 0, stream>>>(stats, 0.f, 128);
        k_bn_stats<<<512, THREADS, 0, stream>>>(A, stats, N_);
        k_bn_apply<<<ewG, THREADS, 0, stream>>>(A, stats, post_g + i * HD, post_beta + i * HD,
                                                N_, i != 2);
        cur = A; float* t = A; A = B; B = t;
    }

    // global mean pool + head
    k_fill<<<(GD * HD + GD + THREADS - 1) / THREADS, THREADS, 0, stream>>>(pooled, 0.f, GD * HD + GD);
    k_pool_accum<<<ewG, THREADS, 0, stream>>>(cur, batch, pooled, counts, N_);
    k_head<<<1, THREADS, 0, stream>>>(pooled, counts, fW0, fb0, fW1, fb1, fW2, fb2, out);
}
